// GAT_28767690949394
// MI455X (gfx1250) — matrix operand order repaired and checked by reading, not compiled, not run on hardware
//
#include <hip/hip_runtime.h>
#include <hip/hip_bf16.h>

#define NN   8192
#define FIN  128
#define FOUT 64
#define NH   4
#define ALPHA 0.2f
#define LOG2E_F 1.4426950408889634f

typedef __bf16 bf16_t;
typedef __attribute__((ext_vector_type(16))) __bf16 v16bf;
typedef __attribute__((ext_vector_type(8)))  float  v8f;
typedef __attribute__((ext_vector_type(4)))  float  v4f;

// ---------------------------------------------------------------------------
// Kernel 1: per-head projection Wh = x @ W[h], stored pre-swizzled in the
// CDNA5 WMMA 16-bit B-matrix lane layout (32x16 K x N tiles), plus
// f1 = (Wh.a1)*log2e and f2 = (Wh.a2)*log2e  (base-2 softmax domain).
// B-tile element (k,n):  lane = n + ((k/8)%2)*16 , half = k%8 + (k/16)*8
// ---------------------------------------------------------------------------
__global__ __launch_bounds__(256) void gat_proj(
    const float* __restrict__ x, const float* __restrict__ W,
    const float* __restrict__ a1, const float* __restrict__ a2,
    unsigned short* __restrict__ WhB,   // bf16 bits, swizzled
    float* __restrict__ f1s, float* __restrict__ f2s) {
  const int n = blockIdx.x;
  const int t = threadIdx.x;
  __shared__ float xs[FIN];
  __shared__ float red1[256];
  __shared__ float red2[256];
  if (t < FIN) xs[t] = x[(size_t)n * FIN + t];
  __syncthreads();

  const int h = t >> 6;        // head
  const int o = t & 63;        // output feature
  const float* Wp = W + ((size_t)h * FIN) * FOUT + o;
  float acc = 0.f;
#pragma unroll 8
  for (int f = 0; f < FIN; ++f) acc = fmaf(xs[f], Wp[(size_t)f * FOUT], acc);

  // swizzled bf16 store (B-matrix layout for v_wmma_f32_16x16x32_bf16)
  const int jtile = n >> 5, k = n & 31;
  const int ntile = o >> 4, nn = o & 15;
  const int lane  = nn + (((k >> 3) & 1) << 4);
  const int half  = (k & 7) | ((k & 16) >> 1);
  const size_t idx =
      ((size_t)((h * (NN / 32) + jtile) * 4 + ntile)) * 512 + (size_t)lane * 16 + half;
  union { bf16_t b; unsigned short u; } cv;
  cv.b = (bf16_t)acc;
  WhB[idx] = cv.u;

  red1[t] = acc * a1[h * FOUT + o];
  red2[t] = acc * a2[h * FOUT + o];
  __syncthreads();
  if (o == 0) {
    float s1 = 0.f, s2 = 0.f;
#pragma unroll 8
    for (int i = 0; i < 64; ++i) { s1 += red1[t + i]; s2 += red2[t + i]; }
    f1s[(size_t)h * NN + n] = s1 * LOG2E_F;
    f2s[(size_t)h * NN + n] = s2 * LOG2E_F;
  }
}

// ---------------------------------------------------------------------------
// Kernel 1b: replicate f2 into the A-fragment lane/half swizzle so the
// attention kernel loads it with coalesced b128s instead of 16 bpermutes.
// f2sw[h][jtile][lane][hi] = f2s[h][jtile*32 + K(hi,lane)]
//   K(hi,lane) = (hi&7) + (hi>>3)*16 + (lane>>4)*8
// ---------------------------------------------------------------------------
__global__ __launch_bounds__(256) void gat_f2swz(
    const float* __restrict__ f2s, float* __restrict__ f2sw) {
  const int idx = blockIdx.x * blockDim.x + threadIdx.x;   // over H*NN
  const int h = idx >> 13;        // / NN
  const int n = idx & (NN - 1);
  const float v = f2s[idx];
  const int jt  = n >> 5, k = n & 31;
  const int lhi = (k >> 3) & 1;                 // lane group holding this k
  const int hi  = (k & 7) | ((k & 16) >> 1);    // half index holding this k
  float* base = f2sw + (((size_t)h * (NN / 32) + jt) * 32) * 16;
#pragma unroll
  for (int lo = 0; lo < 16; ++lo)               // replicate across the 16 rows
    base[(size_t)(lhi * 16 + lo) * 16 + hi] = v;
}

// ---------------------------------------------------------------------------
// Kernel 1c: per-head global max of f2 (scaled domain). Since leaky-relu is
// monotone increasing, max_j leaky(f1_i + f2_j) = leaky(f1_i + f2max), so the
// softmax shift becomes a loop-invariant constant -- no online max machinery.
// ---------------------------------------------------------------------------
__global__ __launch_bounds__(256) void gat_f2max(
    const float* __restrict__ f2s, float* __restrict__ f2maxg) {
  __shared__ float red[256];
  const float* p = f2s + (size_t)blockIdx.x * NN;   // blockIdx.x = head
  float mx = -3.0e38f;
  for (int i = threadIdx.x; i < NN; i += 256) mx = fmaxf(mx, p[i]);
  red[threadIdx.x] = mx;
  __syncthreads();
  for (int s = 128; s > 0; s >>= 1) {
    if (threadIdx.x < s) red[threadIdx.x] = fmaxf(red[threadIdx.x], red[threadIdx.x + s]);
    __syncthreads();
  }
  if (threadIdx.x == 0) f2maxg[blockIdx.x] = red[0];
}

// ---------------------------------------------------------------------------
// Kernel 2: fused masked-softmax attention + aggregation.
// Block = 128 threads = 4 wave32s; wave w handles head w for a 16-row tile.
// adj tile (16x128 f32) double-buffered in LDS (non-temporal: adj streams
// through exactly once). Closed-form softmax shift (no online max, no
// rescale), P -> bf16 A-fragment; 4 feature WMMAs + 1 all-ones WMMA that
// accumulates the softmax denominator on the XDL pipe.
// ---------------------------------------------------------------------------
__global__ __launch_bounds__(128) void gat_attn(
    const float* __restrict__ adj,
    const unsigned short* __restrict__ WhB,
    const float* __restrict__ f1s, const float* __restrict__ f2sw,
    const float* __restrict__ f2maxg,
    float* __restrict__ out) {
  const int i0   = blockIdx.x << 4;   // 16-row tile base
  const int tid  = threadIdx.x;
  const int h    = tid >> 5;          // wave index == head
  const int lane = tid & 31;
  const int row  = lane & 15;         // A-fragment row for this lane
  const int lhi  = lane >> 4;         // 0: K 0-7/16-23 , 1: K 8-15/24-31

  __shared__ float adjS[2][16 * 132]; // row stride 132 -> conflict-free

  const float f1r  = f1s[(size_t)h * NN + i0 + row];
  const float s0   = f1r + f2maxg[h];
  const float m    = fmaxf(s0, ALPHA * s0);  // == row max of leaky (scaled dom)
  const float f1rm = f1r - m;
  const float negm = -m;

  const unsigned short* WhBh = WhB + (size_t)h * (NN / 32) * 4 * 512;
  const float* f2h = f2sw + (size_t)h * (NN / 32) * 512;

  v8f acc[5] = {};   // [0..3] feature N-tiles, [4] row-sum (softmax denom)

  union { v16bf v; bf16_t e[16]; } bones;  // all-ones B tile (registers only)
#pragma unroll
  for (int i = 0; i < 16; ++i) bones.e[i] = (bf16_t)1.0f;

  // cooperative 16x128 adj tile load (non-temporal: adj is stream-once)
  auto loadChunk = [&](int jc, int buf) {
    const int r  = tid >> 3;
    const int sg = tid & 7;
    const float* src = adj + (size_t)(i0 + r) * NN + jc + sg * 16;
    float* dst = &adjS[buf][r * 132 + sg * 16];
#pragma unroll
    for (int q = 0; q < 4; ++q) {
      const v4f v = __builtin_nontemporal_load((const v4f*)(src + q * 4));
      *(v4f*)(dst + q * 4) = v;
    }
  };

  loadChunk(0, 0);
  __syncthreads();

  for (int jc = 0; jc < NN; jc += 128) {
    const int buf = (jc >> 7) & 1;
    if (jc + 128 < NN) loadChunk(jc + 128, buf ^ 1);
    if (jc + 256 < NN) {   // global_prefetch_b8 two stages ahead
      const int r = tid >> 3, sg = tid & 7;
      __builtin_prefetch(adj + (size_t)(i0 + r) * NN + (jc + 256) + sg * 16, 0, 0);
    }

#pragma unroll
    for (int ks = 0; ks < 4; ++ks) {
      const int jt = (jc >> 5) + ks;

      // f2 in A-layout: 16 contiguous floats per lane (coalesced b128 x4)
      const float* f2p = f2h + ((size_t)jt * 32 + lane) * 16;
      float f2v[16];
#pragma unroll
      for (int q = 0; q < 4; ++q) {
        const v4f fv = *(const v4f*)(f2p + q * 4);
        f2v[q * 4 + 0] = fv[0]; f2v[q * 4 + 1] = fv[1];
        f2v[q * 4 + 2] = fv[2]; f2v[q * 4 + 3] = fv[3];
      }

      // adj in A-layout: two contiguous runs of 8 -> 4x ds_load_b128
      const float* arow = &adjS[buf][row * 132 + ks * 32 + lhi * 8];
      float av[16];
#pragma unroll
      for (int q = 0; q < 4; ++q) {
        const int off = (q >> 1) * 16 + (q & 1) * 4;   // 0,4,16,20
        const v4f a4 = *(const v4f*)(arow + off);
        av[q * 4 + 0] = a4[0]; av[q * 4 + 1] = a4[1];
        av[q * 4 + 2] = a4[2]; av[q * 4 + 3] = a4[3];
      }

      // p = 2^(leaky(f1+f2) - m) * adj ; m is the exact unmasked row max
      union { v16bf v; bf16_t e[16]; } pa;
#pragma unroll
      for (int hi = 0; hi < 16; ++hi) {
        const float e0 = f1r + f2v[hi];
        const float a  = fmaf(ALPHA, e0, negm);   // alpha*e0 - m
        const float b  = f1rm + f2v[hi];          // e0 - m
        const float p  = __builtin_amdgcn_exp2f(fmaxf(a, b)) * av[hi];
        pa.e[hi] = (bf16_t)p;
      }

      const int jt4 = jt << 2;
#pragma unroll
      for (int nt = 0; nt < 4; ++nt) {
        const v16bf b =
            *(const v16bf*)(WhBh + ((size_t)(jt4 + nt)) * 512 + lane * 16);
        acc[nt] = __builtin_amdgcn_wmma_f32_16x16x32_bf16(
            false, pa.v, false, b, (short)0, acc[nt], false, false);
      }
      // softmax denominator: row sums via all-ones B on the matrix pipe
      acc[4] = __builtin_amdgcn_wmma_f32_16x16x32_bf16(
          false, pa.v, false, bones.v, (short)0, acc[4], false, false);
    }
    __syncthreads();
  }

  // epilogue: l lives in every column of acc[4]; rows r = lhi*8 + t sit in
  // lane lhi*16, VGPR t.  Normalize, ELU, write out[node][h*64 + o].
  float lr[8];
#pragma unroll
  for (int t = 0; t < 8; ++t) lr[t] = __shfl(acc[4][t], lhi << 4, 32);
#pragma unroll
  for (int nt = 0; nt < 4; ++nt) {
#pragma unroll
    for (int t = 0; t < 8; ++t) {
      float v = acc[nt][t] / lr[t];
      v = (v > 0.f) ? v : (__expf(v) - 1.f);   // elu
      const int node = i0 + (lhi << 3) + t;
      out[(size_t)node * (NH * FOUT) + h * FOUT + (nt << 4) + row] = v;
    }
  }
}

// ---------------------------------------------------------------------------
extern "C" void kernel_launch(void* const* d_in, const int* in_sizes, int n_in,
                              void* d_out, int out_size, void* d_ws, size_t ws_size,
                              hipStream_t stream) {
  (void)in_sizes; (void)n_in; (void)out_size; (void)ws_size;
  const float* x   = (const float*)d_in[0];   // [N, 128]
  const float* adj = (const float*)d_in[1];   // [N, N]
  const float* W   = (const float*)d_in[2];   // [H, 128, 64]
  const float* a1  = (const float*)d_in[3];   // [H, 64]
  const float* a2  = (const float*)d_in[4];   // [H, 64]
  float* out = (float*)d_out;                 // [N, H*64]

  // workspace: 4MB swizzled bf16 Wh | f1s 128KB | f2s 128KB | f2sw 2MB | f2max
  unsigned short* WhB = (unsigned short*)d_ws;
  float* f1s   = (float*)((char*)d_ws + (size_t)NH * NN * FOUT * sizeof(unsigned short));
  float* f2s   = f1s + (size_t)NH * NN;
  float* f2sw  = f2s + (size_t)NH * NN;
  float* f2mx  = f2sw + (size_t)NH * (NN / 32) * 512;

  gat_proj<<<NN, 256, 0, stream>>>(x, W, a1, a2, WhB, f1s, f2s);
  gat_f2swz<<<(NH * NN) / 256, 256, 0, stream>>>(f2s, f2sw);
  gat_f2max<<<NH, 256, 0, stream>>>(f2s, f2mx);
  gat_attn<<<NN / 16, 128, 0, stream>>>(adj, WhB, f1s, f2sw, f2mx, out);
}
